// MultiHeadAttention_37864431681773
// MI455X (gfx1250) — compile-verified, matrix operand order checked by reading, not run on hardware
//
#include <hip/hip_runtime.h>
#include <hip/hip_bf16.h>

// ---------------------------------------------------------------------------
// MHA forward for MI455X (gfx1250, wave32, WMMA bf16 w/ fp32 accumulation)
// B=2, S=2048, H=2048, NH=16, HD=128
// GEMM: 128 threads = 4 waves (2x2); wave tile = (16*TI)x(16*TJ) via
// TIxTJ v_wmma_f32_16x16x32_bf16 accumulators -> 32 FLOP/B at TI=TJ=4.
// ---------------------------------------------------------------------------

typedef __attribute__((ext_vector_type(16))) __bf16 bf16x16;
typedef __attribute__((ext_vector_type(8)))  __bf16 bf16x8;
typedef __attribute__((ext_vector_type(4)))  __bf16 bf16x4;
typedef __attribute__((ext_vector_type(8)))  float  f32x8;

#define MHA_B   2
#define MHA_S   2048
#define MHA_H   2048
#define MHA_NH  16
#define MHA_HD  128
#define MHA_SCALE 0.08838834764831845f  // 1/sqrt(128)

// ---------------------------------------------------------------------------
// fp32 -> bf16 conversion, vectorized x4
// ---------------------------------------------------------------------------
__global__ void cvt_f32_to_bf16(const float* __restrict__ in,
                                __bf16* __restrict__ out, int n4) {
    int i = blockIdx.x * blockDim.x + threadIdx.x;
    if (i >= n4) return;
    float4 v = ((const float4*)in)[i];
    bf16x4 o;
    o[0] = (__bf16)v.x; o[1] = (__bf16)v.y;
    o[2] = (__bf16)v.z; o[3] = (__bf16)v.w;
    ((bf16x4*)out)[i] = o;
}

// ---------------------------------------------------------------------------
// Fragment loader for v_wmma_f32_16x16x32_bf16 (CDNA5 ISA 7.12.2 layouts).
// 16-bit A (16x32 MxK): lane L -> row = L&15, half = L>>4.
//   half=0: K = [0..7] then [16..23]; half=1: K = [8..15] then [24..31]
// B (32x16 KxN) mirrors A (lane -> col). Two contiguous 16B loads per lane.
// ---------------------------------------------------------------------------
__device__ __forceinline__ bf16x16 load_frag_k32(const __bf16* __restrict__ base,
                                                 int idx, int ld, int k0,
                                                 int half) {
    const __bf16* p = base + (size_t)idx * ld + k0;
    bf16x8 lo = *(const bf16x8*)(p + half * 8);
    bf16x8 hi = *(const bf16x8*)(p + 16 + half * 8);
    return __builtin_shufflevector(lo, hi, 0, 1, 2, 3, 4, 5, 6, 7,
                                   8, 9, 10, 11, 12, 13, 14, 15);
}

// ---------------------------------------------------------------------------
// WMMA GEMM: C[M,N] = A[M,K](bf16) * Bsrc[N,K]^T(bf16) (+ bias[N])
//   - A row-major (lda); Bsrc row-major, row n = column n of B (ldb)
//     -> covers X*W^T and Q*K^T directly. K multiple of 32.
//   - block tile = (32*TI) x (32*TJ); grid = (M/(32*TI), N/(32*TJ)).
//   - TRANS_OUT=true scatters the V-projection into [B,NH,HD,S] layout.
// ---------------------------------------------------------------------------
template <typename TOut, bool TRANS_OUT, int TI, int TJ>
__global__ __launch_bounds__(128) void wmma_gemm_xwt(
    const __bf16* __restrict__ A, int lda,
    const __bf16* __restrict__ Bm, int ldb,
    TOut* __restrict__ C, int ldc,
    const float* __restrict__ bias, int K) {
    const int wave = threadIdx.x >> 5;
    const int lane = threadIdx.x & 31;
    const int half = lane >> 4;
    const int lidx = lane & 15;
    const int row0 = blockIdx.x * (32 * TI) + (wave >> 1) * (16 * TI);
    const int col0 = blockIdx.y * (32 * TJ) + (wave & 1) * (16 * TJ);

    f32x8 acc[TI][TJ] = {};

    for (int k0 = 0; k0 < K; k0 += 32) {
        bf16x16 af[TI], bfr[TJ];
#pragma unroll
        for (int i = 0; i < TI; ++i)
            af[i] = load_frag_k32(A, row0 + 16 * i + lidx, lda, k0, half);
#pragma unroll
        for (int j = 0; j < TJ; ++j)
            bfr[j] = load_frag_k32(Bm, col0 + 16 * j + lidx, ldb, k0, half);
#pragma unroll
        for (int i = 0; i < TI; ++i)
#pragma unroll
            for (int j = 0; j < TJ; ++j)
                acc[i][j] = __builtin_amdgcn_wmma_f32_16x16x32_bf16(
                    false, af[i], false, bfr[j], (short)0, acc[i][j],
                    false, false);
    }

    const bool hasb = (bias != nullptr);
#pragma unroll
    for (int i = 0; i < TI; ++i) {
#pragma unroll
        for (int j = 0; j < TJ; ++j) {
            const int col = col0 + 16 * j + lidx;
            const float bv = hasb ? bias[col] : 0.0f;
#pragma unroll
            for (int r = 0; r < 8; ++r) {
                // C/D layout: VGPR r holds M = r + 8*half, N = lane&15
                const int row = row0 + 16 * i + 8 * half + r;
                const float v = acc[i][j][r] + bv;
                if (!TRANS_OUT) {
                    C[(size_t)row * ldc + col] = (TOut)v;
                } else {
                    // row = b*S + s, col = h*HD + d  ->  [B,NH,HD,S]:
                    // dest = (b*H + col)*S + s
                    const int b = row >> 11;      // S = 2048
                    const int s = row & 2047;
                    C[((size_t)(b * MHA_H + col) << 11) + s] = (TOut)v;
                }
            }
        }
    }
}

// ---------------------------------------------------------------------------
// Row softmax: P[row,:] = softmax(scale * S[row,:]) as bf16. One block/row.
// ---------------------------------------------------------------------------
__global__ __launch_bounds__(256) void softmax_rows(
    const float* __restrict__ Sm, __bf16* __restrict__ P, int n, float scale) {
    __shared__ float red[256];
    const int row = blockIdx.x;
    const float* src = Sm + (size_t)row * n;
    __bf16* dst = P + (size_t)row * n;

    float lmax = -3.0e38f;
    for (int i = threadIdx.x; i < n; i += 256)
        lmax = fmaxf(lmax, src[i] * scale);
    red[threadIdx.x] = lmax;
    __syncthreads();
    for (int s = 128; s > 0; s >>= 1) {
        if (threadIdx.x < s)
            red[threadIdx.x] = fmaxf(red[threadIdx.x], red[threadIdx.x + s]);
        __syncthreads();
    }
    const float m = red[0];
    __syncthreads();

    float lsum = 0.0f;
    for (int i = threadIdx.x; i < n; i += 256)
        lsum += __expf(src[i] * scale - m);
    red[threadIdx.x] = lsum;
    __syncthreads();
    for (int s = 128; s > 0; s >>= 1) {
        if (threadIdx.x < s) red[threadIdx.x] += red[threadIdx.x + s];
        __syncthreads();
    }
    const float inv = 1.0f / red[0];

    for (int i = threadIdx.x; i < n; i += 256)
        dst[i] = (__bf16)(__expf(src[i] * scale - m) * inv);
}

// ---------------------------------------------------------------------------
// Orchestration
// ---------------------------------------------------------------------------
extern "C" void kernel_launch(void* const* d_in, const int* in_sizes, int n_in,
                              void* d_out, int out_size, void* d_ws,
                              size_t ws_size, hipStream_t stream) {
    (void)in_sizes; (void)n_in; (void)out_size; (void)ws_size;

    const int N = MHA_B * MHA_S;           // 4096 rows
    const size_t XE = (size_t)N * MHA_H;   // activation elems (8.4M)
    const size_t WE = (size_t)MHA_H * MHA_H;

    const float* q_in = (const float*)d_in[0];
    const float* k_in = (const float*)d_in[1];
    const float* v_in = (const float*)d_in[2];
    const float* Wq = (const float*)d_in[3];
    const float* bq = (const float*)d_in[4];
    const float* Wk = (const float*)d_in[5];
    const float* bk = (const float*)d_in[6];
    const float* Wv = (const float*)d_in[7];
    const float* bv = (const float*)d_in[8];
    const float* Wo = (const float*)d_in[9];
    const float* bo = (const float*)d_in[10];
    float* out = (float*)d_out;

    // workspace partition (~112 MB total)
    char* wp = (char*)d_ws;
    __bf16* Qb  = (__bf16*)wp; wp += XE * 2;              // 16 MB
    __bf16* Kb  = (__bf16*)wp; wp += XE * 2;              // 16 MB
    __bf16* Vt  = (__bf16*)wp; wp += XE * 2;              // 16 MB [B,NH,HD,S]
    __bf16* Ctx = (__bf16*)wp; wp += XE * 2;              // 16 MB
    __bf16* Xb  = (__bf16*)wp; wp += XE * 2;              // 16 MB (cvt scratch)
    __bf16* Wb  = (__bf16*)wp; wp += WE * 2;              //  8 MB (cvt scratch)
    float*  Sb  = (float*)wp;  wp += (size_t)MHA_S * MHA_S * 4;  // 16 MB
    __bf16* Pb  = (__bf16*)wp; wp += (size_t)MHA_S * MHA_S * 2;  //  8 MB

    const dim3 blk(128);
    const dim3 gProj(N / 128, MHA_H / 128);      // (32, 16), 128x128 blocks
    const int cvtX = (int)(XE / 4 / 256);        // 8192 blocks
    const int cvtW = (int)(WE / 4 / 256);        // 4096 blocks

    // ---- Q projection: Qb = bf16(query @ Wq^T + bq)
    cvt_f32_to_bf16<<<cvtX, 256, 0, stream>>>(q_in, Xb, (int)(XE / 4));
    cvt_f32_to_bf16<<<cvtW, 256, 0, stream>>>(Wq, Wb, (int)(WE / 4));
    wmma_gemm_xwt<__bf16, false, 4, 4><<<gProj, blk, 0, stream>>>(
        Xb, MHA_H, Wb, MHA_H, Qb, MHA_H, bq, MHA_H);

    // ---- K projection
    cvt_f32_to_bf16<<<cvtX, 256, 0, stream>>>(k_in, Xb, (int)(XE / 4));
    cvt_f32_to_bf16<<<cvtW, 256, 0, stream>>>(Wk, Wb, (int)(WE / 4));
    wmma_gemm_xwt<__bf16, false, 4, 4><<<gProj, blk, 0, stream>>>(
        Xb, MHA_H, Wb, MHA_H, Kb, MHA_H, bk, MHA_H);

    // ---- V projection, stored head-transposed [B,NH,HD,S]
    cvt_f32_to_bf16<<<cvtX, 256, 0, stream>>>(v_in, Xb, (int)(XE / 4));
    cvt_f32_to_bf16<<<cvtW, 256, 0, stream>>>(Wv, Wb, (int)(WE / 4));
    wmma_gemm_xwt<__bf16, true, 4, 4><<<gProj, blk, 0, stream>>>(
        Xb, MHA_H, Wb, MHA_H, Vt, MHA_H, bv, MHA_H);

    // ---- attention per (batch, head)
    for (int b = 0; b < MHA_B; ++b) {
        for (int h = 0; h < MHA_NH; ++h) {
            const __bf16* Qh = Qb + (size_t)b * MHA_S * MHA_H + (size_t)h * MHA_HD;
            const __bf16* Kh = Kb + (size_t)b * MHA_S * MHA_H + (size_t)h * MHA_HD;
            // scores = Q_h @ K_h^T  (unscaled; scale folded into softmax)
            wmma_gemm_xwt<float, false, 4, 4>
                <<<dim3(MHA_S / 128, MHA_S / 128), blk, 0, stream>>>(
                    Qh, MHA_H, Kh, MHA_H, Sb, MHA_S, nullptr, MHA_HD);
            // P = softmax(scale * scores) -> bf16, row-major
            softmax_rows<<<MHA_S, 256, 0, stream>>>(Sb, Pb, MHA_S, MHA_SCALE);
            // ctx_h = P @ V_h  (V_h columns contiguous along keys via Vt);
            // N=128 is small -> 128x64 blocks keep grid.y = 2
            const __bf16* Vh = Vt + ((size_t)b * MHA_NH + h) * MHA_HD * MHA_S;
            __bf16* Ch = Ctx + (size_t)b * MHA_S * MHA_H + (size_t)h * MHA_HD;
            wmma_gemm_xwt<__bf16, false, 4, 2>
                <<<dim3(MHA_S / 128, MHA_HD / 64), blk, 0, stream>>>(
                    Pb, MHA_S, Vh, MHA_S, Ch, MHA_H, nullptr, MHA_S);
        }
    }

    // ---- output projection: out = ctx @ Wo^T + bo (fp32 result)
    cvt_f32_to_bf16<<<cvtW, 256, 0, stream>>>(Wo, Wb, (int)(WE / 4));
    wmma_gemm_xwt<float, false, 4, 4><<<gProj, blk, 0, stream>>>(
        Ctx, MHA_H, Wb, MHA_H, out, MHA_H, bo, MHA_H);
}
